// Backbone_23587960389917
// MI455X (gfx1250) — matrix operand order repaired and checked by reading, not compiled, not run on hardware
//
#include <hip/hip_runtime.h>
#include <hip/hip_bf16.h>
#include <stdint.h>
#include <stddef.h>

// ---------------------------------------------------------------------------
// Types for WMMA fragments (gfx1250, wave32)
// ---------------------------------------------------------------------------
typedef _Float16 half_t;
typedef __attribute__((ext_vector_type(16))) _Float16 v16h;
typedef __attribute__((ext_vector_type(8)))  _Float16 v8h;
typedef __attribute__((ext_vector_type(8)))  float    v8f;
typedef __attribute__((ext_vector_type(4)))  float    v4f;

union Frag16 { v16h v; v8h h[2]; };

// A-matrix 16x32 f16 fragment (ISA 7.12.2): lanes 0-15 row M=lane hold
// K={0..7,16..23}; lanes 16-31 row M=lane-16 hold K={8..15,24..31}.
__device__ __forceinline__ v16h frag_a(const half_t* row, int hi) {
  Frag16 f;
  f.h[0] = *(const v8h*)(row + hi * 8);
  f.h[1] = *(const v8h*)(row + 16 + hi * 8);
  return f.v;
}
// B-matrix 32x16 f16 fragment, loaded from B^T (N-major) LDS tile:
// lanes 0-15: N=lane, K={0..7,16..23} ; lanes 16-31: N=lane-16, K={8..15,24..31}.
__device__ __forceinline__ v16h frag_b(const half_t* row, int hi) {
  Frag16 f;
  f.h[0] = *(const v8h*)(row + hi * 8);
  f.h[1] = *(const v8h*)(row + 16 + hi * 8);
  return f.v;
}

__device__ __forceinline__ v8h cvt8(v4f a, v4f b) {
  v8h h;
  h[0] = (half_t)a[0]; h[1] = (half_t)a[1]; h[2] = (half_t)a[2]; h[3] = (half_t)a[3];
  h[4] = (half_t)b[0]; h[5] = (half_t)b[1]; h[6] = (half_t)b[2]; h[7] = (half_t)b[3];
  return h;
}

// ---------------------------------------------------------------------------
// Generic WMMA GEMM:  C[M,N] = epi( A[M,K] * W[K,N] + bias )
// epi: optional BatchNorm (per-channel N), optional ReLU, optional residual.
// Block: 256 threads (8 waves). Block tile 128x64; each wave owns one 16-row
// strip and all four 16-col tiles (4 x v8f accumulators, 4 WMMA per K-slab).
// Fast path (block-uniform, M/N/K aligned): b128 global loads software-
// pipelined one K-slab ahead in registers, packed f16 LDS staging.
// Slow path: branch-free clamped scalar loads for edge tiles.
// ---------------------------------------------------------------------------
__global__ __launch_bounds__(256) void gemm_wmma(
    const float* __restrict__ A, const float* __restrict__ W,
    const float* __restrict__ bias,
    const float* __restrict__ bn_g, const float* __restrict__ bn_b,
    const float* __restrict__ bn_m, const float* __restrict__ bn_v,
    const float* __restrict__ residual, float* __restrict__ C,
    int M, int N, int K, int relu)
{
  __shared__ alignas(16) half_t As[128][40];  // 128 rows x 32 K (+pad)
  __shared__ alignas(16) half_t Bs[64][40];   // 64 N-rows x 32 K (+pad) == W^T

  const int tid  = threadIdx.x;
  const int lane = tid & 31;
  const int wave = tid >> 5;            // 0..7 -> M sixteen-tile
  const int m0 = blockIdx.y * 128;
  const int n0 = blockIdx.x * 64;
  const int lr = lane & 15;
  const int hi = lane >> 4;

  v8f acc[4] = {};

  const bool fast = (m0 + 128 <= M) && (n0 + 64 <= N) &&
                    ((N & 3) == 0) && ((K & 31) == 0);

  if (fast) {
    // A: thread -> row = tid/2 (0..127), col = (tid%2)*16 ; 16 f32/slab
    // W: thread -> k   = tid/8 (0..31),  n  = (tid%8)*8  ;  8 f32/slab
    const int ar = tid >> 1, ac = (tid & 1) << 4;
    const int wk = tid >> 3, wn = (tid & 7) << 3;
    const float* Abase = A + (size_t)(m0 + ar) * K + ac;
    const float* Wbase = W + (size_t)wk * N + (n0 + wn);

    v4f a0 = *(const v4f*)(Abase + 0);
    v4f a1 = *(const v4f*)(Abase + 4);
    v4f a2 = *(const v4f*)(Abase + 8);
    v4f a3 = *(const v4f*)(Abase + 12);
    v4f w0 = *(const v4f*)(Wbase + 0);
    v4f w1 = *(const v4f*)(Wbase + 4);

    for (int k0 = 0; k0 < K; k0 += 32) {
      *(v8h*)(&As[ar][ac])     = cvt8(a0, a1);
      *(v8h*)(&As[ar][ac + 8]) = cvt8(a2, a3);
      v8h wh = cvt8(w0, w1);
#pragma unroll
      for (int j = 0; j < 8; ++j) Bs[wn + j][wk] = wh[j];
      __syncthreads();

      if (k0 + 32 < K) {              // prefetch next slab (overlaps WMMA)
        const float* An = Abase + (k0 + 32);
        const float* Wn = Wbase + (size_t)(k0 + 32) * N;
        a0 = *(const v4f*)(An + 0);
        a1 = *(const v4f*)(An + 4);
        a2 = *(const v4f*)(An + 8);
        a3 = *(const v4f*)(An + 12);
        w0 = *(const v4f*)(Wn + 0);
        w1 = *(const v4f*)(Wn + 4);
      }

      v16h fa = frag_a(&As[wave * 16 + lr][0], hi);
#pragma unroll
      for (int nt = 0; nt < 4; ++nt) {
        v16h fb = frag_b(&Bs[nt * 16 + lr][0], hi);
        acc[nt] = __builtin_amdgcn_wmma_f32_16x16x32_f16(false, fa, false, fb,
                                                         (short)0, acc[nt],
                                                         false, false);
      }
      __syncthreads();
    }
  } else {
    for (int k0 = 0; k0 < K; k0 += 32) {
      // branch-free clamped scalar loads
#pragma unroll
      for (int it = 0; it < 16; ++it) {           // A: 128x32
        int e  = tid + it * 256;
        int r  = e >> 5;
        int cc = e & 31;
        int gr = m0 + r, gc = k0 + cc;
        bool va = (gr < M) && (gc < K);
        float av = A[va ? ((size_t)gr * K + gc) : 0];
        if (!va) av = 0.0f;
        As[r][cc] = (half_t)av;
      }
#pragma unroll
      for (int it = 0; it < 8; ++it) {            // W^T: 64x32
        int e  = tid + it * 256;
        int r  = e >> 5;
        int cc = e & 31;
        int gn = n0 + r, gc = k0 + cc;
        bool vb = (gc < K) && (gn < N);
        float bv = W[vb ? ((size_t)gc * N + gn) : 0];
        if (!vb) bv = 0.0f;
        Bs[r][cc] = (half_t)bv;
      }
      __syncthreads();

      v16h fa = frag_a(&As[wave * 16 + lr][0], hi);
#pragma unroll
      for (int nt = 0; nt < 4; ++nt) {
        v16h fb = frag_b(&Bs[nt * 16 + lr][0], hi);
        acc[nt] = __builtin_amdgcn_wmma_f32_16x16x32_f16(false, fa, false, fb,
                                                         (short)0, acc[nt],
                                                         false, false);
      }
      __syncthreads();
    }
  }

  // Epilogue. C/D layout: VGPR r -> row = r + 8*(lane/16), col = lane%16.
  const int mh = hi * 8;
#pragma unroll
  for (int nt = 0; nt < 4; ++nt) {
    int n = n0 + nt * 16 + lr;
    if (n >= N) continue;
#pragma unroll
    for (int r = 0; r < 8; ++r) {
      int m = m0 + wave * 16 + mh + r;
      if (m >= M) continue;
      float v = acc[nt][r];
      if (bias) v += bias[n];
      if (bn_g) {
        float s = bn_g[n] * rsqrtf(bn_v[n] + 1e-5f);
        v = (v - bn_m[n]) * s + bn_b[n];
      }
      if (relu) v = fmaxf(v, 0.0f);
      if (residual) v += residual[(size_t)m * N + n];
      C[(size_t)m * N + n] = v;
    }
  }
}

// ---------------------------------------------------------------------------
// Support kernels
// ---------------------------------------------------------------------------
__global__ void extract_xyz_kernel(const float* __restrict__ x,
                                   float* __restrict__ xyz, int total) {
  int t = blockIdx.x * blockDim.x + threadIdx.x;
  if (t >= total) return;
  int r = t / 3, c = t % 3;
  xyz[t] = x[(size_t)r * 5 + c];
}

// Zero-pad a KxN weight into KpadxN (rows >= K are zero).
__global__ void pad_weight_kernel(const float* __restrict__ w, float* __restrict__ wp,
                                  int K, int Kpad, int N, int total) {
  int t = blockIdx.x * blockDim.x + threadIdx.x;
  if (t >= total) return;
  int n = t % N, k = t / N;
  wp[t] = (k < K) ? w[(size_t)k * N + n] : 0.0f;
}

// Farthest point sampling: one block per batch, sequential npoint iterations.
__global__ void fps_kernel(const float* __restrict__ xyz, int* __restrict__ outIdx,
                           int N, int npoint) {
  const int b = blockIdx.x, t = threadIdx.x;
  const float* p = xyz + (size_t)b * N * 3;
  __shared__ float sd[256];
  __shared__ int   si[256];
  __shared__ float cc[3];
  float dist[16];
  const int cnt = (N + 255) / 256;
#pragma unroll
  for (int j = 0; j < 16; ++j) dist[j] = 1e10f;
  int far = 0;
  for (int i = 0; i < npoint; ++i) {
    if (t == 0) {
      outIdx[(size_t)b * npoint + i] = far;
      cc[0] = p[far * 3 + 0]; cc[1] = p[far * 3 + 1]; cc[2] = p[far * 3 + 2];
    }
    __syncthreads();
    float best = -1.0f; int bidx = 0;
    for (int j = 0; j < cnt; ++j) {
      int idx = t + j * 256;
      if (idx < N) {
        float dx = p[idx * 3 + 0] - cc[0];
        float dy = p[idx * 3 + 1] - cc[1];
        float dz = p[idx * 3 + 2] - cc[2];
        float d = dx * dx + dy * dy + dz * dz;
        if (d < dist[j]) dist[j] = d;
        if (dist[j] > best) { best = dist[j]; bidx = idx; }
      }
    }
    sd[t] = best; si[t] = bidx;
    __syncthreads();
    for (int s2 = 128; s2 > 0; s2 >>= 1) {
      if (t < s2 && sd[t + s2] > sd[t]) { sd[t] = sd[t + s2]; si[t] = si[t + s2]; }
      __syncthreads();
    }
    far = si[0];
    __syncthreads();
  }
}

// Brute-force KNN (K=16): one thread per query point.
__global__ void knn_kernel(const float* __restrict__ q, const float* __restrict__ src,
                           int* __restrict__ out, int Nq, int Ns, int total) {
  int t = blockIdx.x * blockDim.x + threadIdx.x;
  if (t >= total) return;
  int b = t / Nq;
  const float* sp = src + (size_t)b * Ns * 3;
  float qx = q[(size_t)t * 3], qy = q[(size_t)t * 3 + 1], qz = q[(size_t)t * 3 + 2];
  float bd[16]; int bi[16];
#pragma unroll
  for (int k = 0; k < 16; ++k) { bd[k] = 3.4e38f; bi[k] = 0; }
  int worst = 0; float wv = 3.4e38f;
  for (int j = 0; j < Ns; ++j) {
    float dx = sp[j * 3 + 0] - qx;
    float dy = sp[j * 3 + 1] - qy;
    float dz = sp[j * 3 + 2] - qz;
    float d = dx * dx + dy * dy + dz * dz;
    if (d < wv) {
      bd[worst] = d; bi[worst] = j;
      worst = 0; wv = bd[0];
#pragma unroll
      for (int k = 1; k < 16; ++k)
        if (bd[k] > wv) { wv = bd[k]; worst = k; }
    }
  }
  for (int a = 0; a < 15; ++a) {          // sort ascending (top_k order)
    int m = a;
    for (int k = a + 1; k < 16; ++k) if (bd[k] < bd[m]) m = k;
    float td = bd[a]; bd[a] = bd[m]; bd[m] = td;
    int   ti = bi[a]; bi[a] = bi[m]; bi[m] = ti;
  }
#pragma unroll
  for (int k = 0; k < 16; ++k) out[(size_t)t * 16 + k] = bi[k];
}

__global__ void gather_rows_kernel(const float* __restrict__ in, const int* __restrict__ idx,
                                   float* __restrict__ out, int Nout, int Nin, int C, int total) {
  int t = blockIdx.x * blockDim.x + threadIdx.x;
  if (t >= total) return;
  int c = t % C, r = t / C, bb = r / Nout;
  out[t] = in[((size_t)bb * Nin + idx[r]) * C + c];
}

// Grouped SA input: concat(grouped_xyz - new_xyz, grouped_points), channel-
// padded to Cpad with zeros so the GEMM can take the vectorized path.
__global__ void build_group_kernel(const float* __restrict__ xyz, const float* __restrict__ nxyz,
                                   const float* __restrict__ pts, const int* __restrict__ knn,
                                   float* __restrict__ out, int np, int Nin, int Cp,
                                   int Cpad, int Kn, int total) {
  int t = blockIdx.x * blockDim.x + threadIdx.x;
  if (t >= total) return;
  int c = t % Cpad;
  int row = t / Cpad;           // (b*np + j)*Kn + k
  int bj = row / Kn;            // b*np + j
  int bb = bj / np;
  int src = knn[row];
  float v = 0.0f;
  if (c < 3)            v = xyz[((size_t)bb * Nin + src) * 3 + c] - nxyz[(size_t)bj * 3 + c];
  else if (c < 3 + Cp)  v = pts[((size_t)bb * Nin + src) * (size_t)Cp + (c - 3)];
  out[t] = v;
}

// pos1 = relu(delta1(xyz_i - xyz_knn)): block per (b,i,k)-row, thread per d.
__global__ void delta1_kernel(const float* __restrict__ xyz, const int* __restrict__ knn,
                              const float* __restrict__ w, const float* __restrict__ b,
                              float* __restrict__ out, int Nq, int Kn) {
  int row = blockIdx.x;          // (b*Nq + i)*Kn + k
  int bi  = row / Kn;
  int bb  = bi / Nq;
  int j   = knn[row];
  __shared__ float dx[3];
  if (threadIdx.x < 3)
    dx[threadIdx.x] = xyz[(size_t)bi * 3 + threadIdx.x] -
                      xyz[((size_t)bb * Nq + j) * 3 + threadIdx.x];
  __syncthreads();
  int d = threadIdx.x;
  float v = dx[0] * w[d] + dx[1] * w[256 + d] + dx[2] * w[512 + d] + b[d];
  out[(size_t)row * 256 + d] = fmaxf(v, 0.0f);
}

// t1 = q - gather(kf) + pos ; pos += gather(vf)   (pos becomes v+pos, in-place)
__global__ void attn_prep_kernel(const float* __restrict__ q, const float* __restrict__ kf,
                                 const float* __restrict__ vf, const int* __restrict__ knn,
                                 float* __restrict__ pos, float* __restrict__ t1,
                                 int Nq, int Kn) {
  int row = blockIdx.x;
  int bi  = row / Kn;
  int bb  = bi / Nq;
  int j   = knn[row];
  size_t gs = ((size_t)bb * Nq + j) * 256 + threadIdx.x;
  size_t qi = (size_t)bi * 256 + threadIdx.x;
  size_t ri = (size_t)row * 256 + threadIdx.x;
  float p = pos[ri];
  t1[ri]  = q[qi] - kf[gs] + p;
  pos[ri] = p + vf[gs];
}

// softmax over K (axis=-2) per (b,i,d) with 1/sqrt(256) scaling, then
// weighted sum with vpp -> res[b,i,d]
__global__ void softmax_ws_kernel(const float* __restrict__ attn, const float* __restrict__ vpp,
                                  float* __restrict__ res, int Kn, int total) {
  int t = blockIdx.x * blockDim.x + threadIdx.x;
  if (t >= total) return;
  int d = t & 255, bi = t >> 8;
  size_t base = ((size_t)bi * Kn) * 256 + d;
  float a[16];
  float m = -3.4e38f;
  for (int k = 0; k < Kn; ++k) {
    a[k] = attn[base + (size_t)k * 256] * 0.0625f;   // /sqrt(256)
    if (a[k] > m) m = a[k];
  }
  float s = 0.0f;
  for (int k = 0; k < Kn; ++k) { a[k] = __expf(a[k] - m); s += a[k]; }
  float inv = 1.0f / s, acc = 0.0f;
  for (int k = 0; k < Kn; ++k) acc += a[k] * vpp[base + (size_t)k * 256];
  res[t] = acc * inv;
}

__global__ void maxpool_k_kernel(const float* __restrict__ in, float* __restrict__ out,
                                 int Kn, int C, int total) {
  int t = blockIdx.x * blockDim.x + threadIdx.x;
  if (t >= total) return;
  int c = t % C, r = t / C;
  float m = -3.4e38f;
  for (int k = 0; k < Kn; ++k)
    m = fmaxf(m, in[((size_t)r * Kn + k) * C + c]);
  out[t] = m;
}

// ---------------------------------------------------------------------------
// Host-side orchestration
// ---------------------------------------------------------------------------
struct LinP { const float* w; const float* b; };
struct BNPr { const float* g; const float* be; const float* mn; const float* vr; };
struct SAP  { BNPr bn0, bn1; LinP c0, c1; };
struct TBP  { LinP fc1, fc2, d1, d2, g1, g2; const float *wq, *wk, *wv; };
struct Scr  { float *h, *q, *kf, *vf, *res, *P1, *P2, *P3; int* knn; };

static inline int cdiv(int a, int b) { return (a + b - 1) / b; }

static void gemm(hipStream_t s, const float* A, const float* W, const float* bias,
                 const BNPr* bn, const float* resid, float* C,
                 int M, int N, int K, int relu) {
  dim3 grid((unsigned)cdiv(N, 64), (unsigned)cdiv(M, 128));
  gemm_wmma<<<grid, 256, 0, s>>>(A, W, bias,
      bn ? bn->g : nullptr, bn ? bn->be : nullptr,
      bn ? bn->mn : nullptr, bn ? bn->vr : nullptr,
      resid, C, M, N, K, relu);
}

static void run_tb(hipStream_t s, const TBP& p, const float* xyz, int Nq, int dp,
                   const float* ptsIn, float* ptsOut, const Scr& w) {
  const int B = 4, Kn = 16;
  const int bi = B * Nq;
  const int R  = bi * Kn;
  knn_kernel<<<cdiv(bi, 128), 128, 0, s>>>(xyz, xyz, w.knn, Nq, Nq, bi);
  gemm(s, ptsIn, p.fc1.w, p.fc1.b, nullptr, nullptr, w.h, bi, 256, dp, 0);
  gemm(s, w.h, p.wq, nullptr, nullptr, nullptr, w.q,  bi, 256, 256, 0);
  gemm(s, w.h, p.wk, nullptr, nullptr, nullptr, w.kf, bi, 256, 256, 0);
  gemm(s, w.h, p.wv, nullptr, nullptr, nullptr, w.vf, bi, 256, 256, 0);
  delta1_kernel<<<R, 256, 0, s>>>(xyz, w.knn, p.d1.w, p.d1.b, w.P1, Nq, Kn);
  gemm(s, w.P1, p.d2.w, p.d2.b, nullptr, nullptr, w.P2, R, 256, 256, 0);  // pos
  attn_prep_kernel<<<R, 256, 0, s>>>(w.q, w.kf, w.vf, w.knn, w.P2, w.P1, Nq, Kn);
  gemm(s, w.P1, p.g1.w, p.g1.b, nullptr, nullptr, w.P3, R, 256, 256, 1);
  gemm(s, w.P3, p.g2.w, p.g2.b, nullptr, nullptr, w.P1, R, 256, 256, 0);  // attn
  softmax_ws_kernel<<<cdiv(bi * 256, 256), 256, 0, s>>>(w.P1, w.P2, w.res, Kn, bi * 256);
  gemm(s, w.res, p.fc2.w, p.fc2.b, nullptr, ptsIn, ptsOut, bi, dp, 256, 0);
}

static void run_sa(hipStream_t s, const SAP& p, const float* xyz, const float* pts,
                   int N, int np, int cprev, int c,
                   float* nxyz, float* npts, int* fpsIdx, int* knn,
                   float* G, float* G1, float* G2, float* wpad) {
  const int B = 4, Kn = 16;
  fps_kernel<<<B, 256, 0, s>>>(xyz, fpsIdx, N, np);
  int tg = B * np * 3;
  gather_rows_kernel<<<cdiv(tg, 256), 256, 0, s>>>(xyz, fpsIdx, nxyz, np, N, 3, tg);
  knn_kernel<<<cdiv(B * np, 128), 128, 0, s>>>(nxyz, xyz, knn, np, N, B * np);
  int Cin  = 3 + cprev;
  int Cpad = (Cin + 31) & ~31;          // 64,96,160,288
  int rowsG = B * np * Kn;
  int tot = rowsG * Cpad;
  build_group_kernel<<<cdiv(tot, 256), 256, 0, s>>>(xyz, nxyz, pts, knn, G,
                                                    np, N, cprev, Cpad, Kn, tot);
  int tw = Cpad * c;
  pad_weight_kernel<<<cdiv(tw, 256), 256, 0, s>>>(p.c0.w, wpad, Cin, Cpad, c, tw);
  gemm(s, G,  wpad,   p.c0.b, &p.bn0, nullptr, G1, rowsG, c, Cpad, 1);
  gemm(s, G1, p.c1.w, p.c1.b, &p.bn1, nullptr, G2, rowsG, c, c,    1);
  int tp = B * np * c;
  maxpool_k_kernel<<<cdiv(tp, 256), 256, 0, s>>>(G2, npts, Kn, c, tp);
}

extern "C" void kernel_launch(void* const* d_in, const int* in_sizes, int n_in,
                              void* d_out, int out_size, void* d_ws, size_t ws_size,
                              hipStream_t stream) {
  auto f = [&](int i) { return (const float*)d_in[i]; };

  const float* X = nullptr;
  LinP fc1a, fc1b;
  SAP sa[4];
  TBP tb[5];   // tb[0] = tb0, tb[1..4] = tb list

  // Two plausible flattening orders of the inputs dict: insertion order
  // (x first) vs jax tree_leaves sorted-key order (x last). Detect via size.
  bool insertion = (in_sizes[0] == 4 * 4096 * 5);
  if (insertion) {
    X = f(0);
    fc1a = { f(1), f(2) };
    fc1b = { f(3), f(4) };
    auto loadTB = [&](int b) {
      TBP t;
      t.fc1 = { f(b + 0),  f(b + 1) };  t.fc2 = { f(b + 2),  f(b + 3) };
      t.d1  = { f(b + 4),  f(b + 5) };  t.d2  = { f(b + 6),  f(b + 7) };
      t.g1  = { f(b + 8),  f(b + 9) };  t.g2  = { f(b + 10), f(b + 11) };
      t.wq = f(b + 12); t.wk = f(b + 13); t.wv = f(b + 14);
      return t;
    };
    tb[0] = loadTB(5);
    for (int i = 0; i < 4; ++i) {
      int b2 = 20 + 12 * i;
      sa[i].c0  = { f(b2 + 0), f(b2 + 1) };
      sa[i].c1  = { f(b2 + 2), f(b2 + 3) };
      sa[i].bn0 = { f(b2 + 4), f(b2 + 5), f(b2 + 6), f(b2 + 7) };
      sa[i].bn1 = { f(b2 + 8), f(b2 + 9), f(b2 + 10), f(b2 + 11) };
    }
    for (int i = 0; i < 4; ++i) tb[1 + i] = loadTB(68 + 15 * i);
  } else {
    X = f(127);
    fc1a = { f(1), f(0) };
    fc1b = { f(3), f(2) };
    auto loadTB = [&](int b) {
      TBP t;
      t.d1  = { f(b + 1),  f(b + 0) };  t.d2  = { f(b + 3),  f(b + 2) };
      t.fc1 = { f(b + 5),  f(b + 4) };  t.fc2 = { f(b + 7),  f(b + 6) };
      t.g1  = { f(b + 9),  f(b + 8) };  t.g2  = { f(b + 11), f(b + 10) };
      t.wk = f(b + 12); t.wq = f(b + 13); t.wv = f(b + 14);
      return t;
    };
    for (int i = 0; i < 4; ++i) {
      int b2 = 4 + 12 * i;
      sa[i].bn0 = { f(b2 + 1), f(b2 + 0), f(b2 + 2), f(b2 + 3) };
      sa[i].bn1 = { f(b2 + 5), f(b2 + 4), f(b2 + 6), f(b2 + 7) };
      sa[i].c0  = { f(b2 + 9),  f(b2 + 8) };
      sa[i].c1  = { f(b2 + 11), f(b2 + 10) };
    }
    for (int i = 0; i < 4; ++i) tb[1 + i] = loadTB(52 + 15 * i);
    tb[0] = loadTB(112);
  }

  // ---- workspace carve (bump allocator, 256B aligned) ----
  size_t off = 0;
  auto alloc = [&](size_t nbytes) -> void* {
    void* p = (char*)d_ws + off;
    off += (nbytes + 255) & ~(size_t)255;
    return p;
  };
  const size_t F = sizeof(float);
  float* xyzA = (float*)alloc((size_t)4 * 4096 * 3 * F);
  float* xyzB = (float*)alloc((size_t)4 * 4096 * 3 * F);
  float* ptsA = (float*)alloc((size_t)4 * 4096 * 64 * F);
  float* ptsB = (float*)alloc((size_t)4 * 4096 * 64 * F);
  Scr scr;
  scr.h   = (float*)alloc((size_t)16384 * 256 * F);
  scr.q   = (float*)alloc((size_t)16384 * 256 * F);
  scr.kf  = (float*)alloc((size_t)16384 * 256 * F);
  scr.vf  = (float*)alloc((size_t)16384 * 256 * F);
  scr.res = (float*)alloc((size_t)16384 * 256 * F);
  scr.P1  = (float*)alloc((size_t)262144 * 256 * F);
  scr.P2  = (float*)alloc((size_t)262144 * 256 * F);
  scr.P3  = (float*)alloc((size_t)262144 * 256 * F);
  scr.knn = (int*)alloc((size_t)262144 * sizeof(int));
  int* fpsIdx = (int*)alloc((size_t)4096 * sizeof(int));
  float* G    = (float*)alloc((size_t)65536 * 64 * F);   // rowsG x Cpad (max 65536x64)
  float* G1   = (float*)alloc((size_t)65536 * 64 * F);
  float* G2   = (float*)alloc((size_t)65536 * 64 * F);
  float* wpad = (float*)alloc((size_t)288 * 512 * F);    // padded conv0 weights
  (void)ws_size; (void)n_in; (void)out_size;

  // ---- forward pass ----
  extract_xyz_kernel<<<cdiv(4 * 4096 * 3, 256), 256, 0, stream>>>(X, xyzA, 4 * 4096 * 3);
  // feats = fc1b(relu(fc1a(x)))   (A = x viewed as (16384,5))
  gemm(stream, X, fc1a.w, fc1a.b, nullptr, nullptr, scr.h, 16384, 32, 5, 1);
  gemm(stream, scr.h, fc1b.w, fc1b.b, nullptr, nullptr, ptsA, 16384, 32, 32, 0);

  float* xyz  = xyzA; float* xyz2 = xyzB;
  float* pts  = ptsA; float* pts2 = ptsB;

  run_tb(stream, tb[0], xyz, 4096, 32, pts, pts, scr);

  int N = 4096;
  for (int i = 0; i < 4; ++i) {
    int np = N / 4;
    int c = 64 << i;          // 64,128,256,512
    int cprev = 32 << i;      // 32,64,128,256
    run_sa(stream, sa[i], xyz, pts, N, np, cprev, c,
           xyz2, pts2, fpsIdx, scr.knn, G, G1, G2, wpad);
    { float* t = xyz; xyz = xyz2; xyz2 = t; }
    { float* t = pts; pts = pts2; pts2 = t; }
    float* outp = (i == 3) ? (float*)d_out : pts;
    run_tb(stream, tb[1 + i], xyz, np, c, pts, outp, scr);
    N = np;
  }
}
